// Sketch3ResidualBlock_15298673508736
// MI455X (gfx1250) — matrix operand order repaired and checked by reading, not compiled, not run on hardware
//
#include <hip/hip_runtime.h>
#include <hip/hip_bf16.h>
#include <stdint.h>

// Sketch3ResidualBlock for MI455X (gfx1250).
// Live path only (out1 is dead in the reference):
//   out = relu(BN(median_j(unsketch_j(conv(x, sketch_j(w2)))))) + x
// conv done as implicit GEMM with v_wmma_f32_16x16x32_bf16, weights streamed
// to LDS with TDM (tensor_load_to_lds + s_wait_tensorcnt double buffering).

#define B_    32
#define C_    128
#define H_    64
#define W_    64
#define D_    32
#define J_    3
#define JD_   96          // J_*D_ conv output channels
#define KTOT  1152        // 9 taps * 128 input channels (tap-major K order)
#define CHUNK 32          // K per WMMA
#define NCHUNK 36         // KTOT / CHUNK

typedef __attribute__((ext_vector_type(16))) __bf16 v16bf;
typedef __attribute__((ext_vector_type(8)))  float  v8f;
typedef __attribute__((ext_vector_type(4)))  unsigned int u32x4;
typedef __attribute__((ext_vector_type(8)))  int i32x8;
typedef __attribute__((ext_vector_type(4)))  int i32x4;

union FragAB {
  v16bf v;
  uint4 q[2];
};

__device__ __forceinline__ unsigned short f2bf(float f) {
  unsigned u = __float_as_uint(f);
  unsigned r = u + 0x7FFFu + ((u >> 16) & 1u);   // round-to-nearest-even
  return (unsigned short)(r >> 16);
}

__device__ __forceinline__ unsigned lds_off(const void* p) {
  // generic pointer to LDS: low 32 bits are the LDS byte offset
  return (unsigned)(unsigned long long)(uintptr_t)p;
}

// ---------------------------------------------------------------- prep ------
__global__ void k_prep_small(const float* __restrict__ hs,
                             float* gsum, float* gsumsq, int* idx) {
  int t = threadIdx.x;
  if (t < C_) { gsum[t] = 0.f; gsumsq[t] = 0.f; }
  if (t < J_ * C_) {
    int j = t / C_, c = t - (t / C_) * C_;
    int d_found = 0;
    for (int d = 0; d < D_; ++d)
      if (hs[j * D_ * C_ + d * C_ + c] > 0.5f) d_found = d;
    idx[t] = d_found;
  }
}

// sketched weights, bf16, row-major [96][KTOT] with tap-major K: k = tap*128+ci
__global__ void k_prep_w(const float* __restrict__ w2,
                         const float* __restrict__ hs,
                         const float* __restrict__ ss,
                         unsigned short* __restrict__ w2sk) {
  int t = blockIdx.x * blockDim.x + threadIdx.x;
  if (t >= JD_ * KTOT) return;
  int row = t / KTOT;            // j*32 + d
  int k   = t - row * KTOT;      // tap*128 + ci
  int j = row >> 5, d = row & 31;
  int tap = k >> 7, ci = k & 127;
  float acc = 0.f;
  for (int c = 0; c < C_; ++c) {
    float hval = hs[j * D_ * C_ + d * C_ + c];     // one-hot
    acc += hval * ss[j * C_ + c] * w2[(c * C_ + ci) * 9 + tap];
  }
  w2sk[t] = f2bf(acc);
}

// --------------------------------------------------------------- TDM --------
__device__ __forceinline__ void tdm_load_chunk(const unsigned short* w2sk,
                                               int kkg, unsigned lds_byte) {
  unsigned long long ga = (unsigned long long)(uintptr_t)(w2sk + kkg * CHUNK);
  u32x4 g0;
  g0[0] = 1u;                                        // count=1, user descriptor
  g0[1] = lds_byte;                                  // lds_addr
  g0[2] = (unsigned)(ga & 0xFFFFFFFFull);            // global_addr[31:0]
  g0[3] = (unsigned)((ga >> 32) & 0x1FFFFFFull) | (2u << 30);  // ga[56:32]|type=2
  i32x8 g1;
  g1[0] = (int)(1u << 16);                           // data_size=1 (2 bytes)
  g1[1] = (int)(((unsigned)KTOT & 0xFFFFu) << 16);   // tensor_dim0 lo16
  g1[2] = (int)(((unsigned)KTOT >> 16) | ((unsigned)JD_ << 16)); // td0 hi|td1 lo
  g1[3] = (int)((unsigned)CHUNK << 16);              // td1 hi=0 | tile_dim0=32
  g1[4] = (int)JD_;                                  // tile_dim1=96, tile_dim2=0
  g1[5] = (int)KTOT;                                 // tensor_dim0_stride lo32
  g1[6] = 0;
  g1[7] = 0;
  i32x4 z4; z4[0] = 0; z4[1] = 0; z4[2] = 0; z4[3] = 0;
  i32x8 z8;
#pragma unroll
  for (int e = 0; e < 8; ++e) z8[e] = 0;
  // 6-arg form (clang-23 / therock-10.0 headers): groups 0..3 + extra + cpol
  __builtin_amdgcn_tensor_load_to_lds(g0, g1, z4, z4, z8, 0);
}

// --------------------------------------------------------------- conv -------
// grid: 32*64 blocks (one per (b,h)); 128 threads = 4 waves.
// wave w owns N-tile (pixels w*16..w*16+15) and all six 16-row M-tiles.
__global__ __launch_bounds__(128) void k_conv(
    const float* __restrict__ x, const unsigned short* __restrict__ w2sk,
    const int* __restrict__ idx, const float* __restrict__ ss,
    float* __restrict__ ymed, float* __restrict__ gsum,
    float* __restrict__ gsumsq) {
  __shared__ unsigned short xs[3 * 66 * 64];        // 25,344 B (bf16 slab+halo)
  __shared__ unsigned short wbuf[2][JD_ * CHUNK];   // 12,288 B (TDM dbl buf)
  __shared__ float ytile[JD_ * W_];                 // 24,576 B

  const int tid  = threadIdx.x;
  const int lane = tid & 31;
  const int wid  = tid >> 5;
  const int b    = blockIdx.x >> 6;
  const int h    = blockIdx.x & 63;

  const int n0   = wid * 16;
  const int mrow = lane & 15;
  const int klo  = (lane < 16) ? 0 : 16;   // A-frag: contiguous 16 K per lane
  const int blo  = (lane < 16) ? 0 : 16;   // B-frag: contiguous 16 K per lane
  const int wN   = n0 + mrow;              // this lane's pixel / B column

  v8f acc[6];
#pragma unroll
  for (int mt = 0; mt < 6; ++mt)
#pragma unroll
    for (int e = 0; e < 8; ++e) acc[mt][e] = 0.f;

  if (wid == 0) {                          // preload first two weight chunks
    tdm_load_chunk(w2sk, 0, lds_off(&wbuf[0][0]));
    tdm_load_chunk(w2sk, 1, lds_off(&wbuf[1][0]));
  }

  for (int hf = 0; hf < 2; ++hf) {         // two 64-channel halves of K
    // stage x[b, hf*64 + ci, h-1..h+1, -1..64] as bf16 (coalesced in w)
    for (int i = tid; i < 3 * 66 * 64; i += 128) {
      int col = i % 66;
      int rc  = i / 66;
      int ci  = rc & 63;
      int r   = rc >> 6;
      int hh  = h + r - 1;
      int ww  = col - 1;
      float v = 0.f;
      if (hh >= 0 && hh < H_ && ww >= 0 && ww < W_)
        v = x[(((b * C_) + hf * 64 + ci) * H_ + hh) * W_ + ww];
      xs[(r * 66 + col) * 64 + ci] = f2bf(v);
    }
    __syncthreads();

    for (int kk = 0; kk < 18; ++kk) {
      const int kkg = hf * 18 + kk;
      if (wid == 0) {                      // chunk kkg must have landed
        if (kkg == NCHUNK - 1) __builtin_amdgcn_s_wait_tensorcnt(0);
        else                   __builtin_amdgcn_s_wait_tensorcnt(1);
      }
      __syncthreads();

      const int tap = kk >> 1;
      const int kh  = tap / 3, kw = tap - kh * 3;
      const int cib = (kk & 1) * 32;       // which 32 of the 64 channels

      // B fragment: patch column for pixel wN, K = cib+blo .. +15 contiguous
      const unsigned short* bsrc = &xs[((kh * 66) + wN + kw) * 64 + cib + blo];
      FragAB bfrag;
      bfrag.q[0] = *(const uint4*)(bsrc);
      bfrag.q[1] = *(const uint4*)(bsrc + 8);

      const unsigned short* wrow = &wbuf[kkg & 1][0];
#pragma unroll
      for (int mt = 0; mt < 6; ++mt) {
        // A fragment: weight row mt*16+mrow, K runs klo..klo+15 contiguous
        const unsigned short* asrc = wrow + (mt * 16 + mrow) * CHUNK + klo;
        FragAB afrag;
        afrag.q[0] = *(const uint4*)(asrc);
        afrag.q[1] = *(const uint4*)(asrc + 8);
        acc[mt] = __builtin_amdgcn_wmma_f32_16x16x32_bf16(
            false, afrag.v, false, bfrag.v, (short)0, acc[mt], false, false);
      }
      __syncthreads();                     // all waves done with wbuf[kkg&1]
      if (wid == 0 && kkg + 2 < NCHUNK)
        tdm_load_chunk(w2sk, kkg + 2, lds_off(&wbuf[kkg & 1][0]));
    }
  }

  // spill the 96x64 conv tile to LDS (C/D layout: VGPR r -> M = r(+8), N = lane%16)
#pragma unroll
  for (int mt = 0; mt < 6; ++mt) {
#pragma unroll
    for (int r = 0; r < 8; ++r) {
      int m = mt * 16 + r + ((lane < 16) ? 0 : 8);
      ytile[m * W_ + wN] = acc[mt][r];
    }
  }
  __syncthreads();

  // unsketch (gather) + sign + median-of-3 + BN partial sums; thread = channel
  {
    const int c = tid;                     // 128 threads == 128 channels
    const int r0 = 0 * D_ + idx[0 * C_ + c];
    const int r1 = 1 * D_ + idx[1 * C_ + c];
    const int r2 = 2 * D_ + idx[2 * C_ + c];
    const float s0 = ss[0 * C_ + c];
    const float s1 = ss[1 * C_ + c];
    const float s2 = ss[2 * C_ + c];
    float sum = 0.f, sumsq = 0.f;
    float* orow = &ymed[(((b * C_) + c) * H_ + h) * W_];
    for (int w = 0; w < W_; ++w) {
      float a  = s0 * ytile[r0 * W_ + w];
      float bb = s1 * ytile[r1 * W_ + w];
      float cc = s2 * ytile[r2 * W_ + w];
      float mx = fmaxf(a, fmaxf(bb, cc));
      float mn = fminf(a, fminf(bb, cc));
      float med = a + bb + cc - mx - mn;   // median of 3
      orow[w] = med;
      sum += med;
      sumsq += med * med;
    }
    atomicAdd(&gsum[c], sum);
    atomicAdd(&gsumsq[c], sumsq);
  }
}

// ------------------------------------------------------------ finalize ------
// in-place on d_out (which holds the median y): BN + relu + residual
__global__ void k_finalize(const float* __restrict__ x,
                           const float* __restrict__ gamma,
                           const float* __restrict__ beta,
                           const float* __restrict__ gsum,
                           const float* __restrict__ gsumsq,
                           float* __restrict__ out) {
  int i = blockIdx.x * blockDim.x + threadIdx.x;  // float4 index
  int c = (i >> 10) & 127;                        // 4096 floats per (b,c)
  const float invn = 1.f / 131072.f;              // N*H*W
  float mean = gsum[c] * invn;
  float var  = gsumsq[c] * invn - mean * mean;    // biased, like jnp.var
  float inv  = rsqrtf(var + 1e-5f);
  float g    = gamma[c] * inv;
  float be   = beta[c] - mean * g;
  float4 y  = ((const float4*)out)[i];
  float4 xr = ((const float4*)x)[i];
  float4 r;
  r.x = fmaxf(y.x * g + be, 0.f) + xr.x;
  r.y = fmaxf(y.y * g + be, 0.f) + xr.y;
  r.z = fmaxf(y.z * g + be, 0.f) + xr.z;
  r.w = fmaxf(y.w * g + be, 0.f) + xr.w;
  ((float4*)out)[i] = r;
}

// -------------------------------------------------------------- launch ------
extern "C" void kernel_launch(void* const* d_in, const int* in_sizes, int n_in,
                              void* d_out, int out_size, void* d_ws,
                              size_t ws_size, hipStream_t stream) {
  const float* x      = (const float*)d_in[0];
  const float* w2     = (const float*)d_in[2];   // w1/gamma1/beta1 are dead
  const float* gamma2 = (const float*)d_in[5];
  const float* beta2  = (const float*)d_in[6];
  const float* hs     = (const float*)d_in[7];
  const float* ssp    = (const float*)d_in[8];

  char* ws = (char*)d_ws;
  unsigned short* w2sk = (unsigned short*)ws;              // 221,184 B
  int*   idx    = (int*)(ws + 221184);                     //   1,536 B
  float* gsum   = (float*)(ws + 222720);                   //     512 B
  float* gsumsq = (float*)(ws + 223232);                   //     512 B
  float* out = (float*)d_out;

  k_prep_small<<<1, 384, 0, stream>>>(hs, gsum, gsumsq, idx);
  k_prep_w<<<(JD_ * KTOT + 255) / 256, 256, 0, stream>>>(w2, hs, ssp, w2sk);
  k_conv<<<B_ * H_, 128, 0, stream>>>(x, w2sk, idx, ssp, out, gsum, gsumsq);
  k_finalize<<<(B_ * C_ * H_ * W_ / 4) / 256, 256, 0, stream>>>(
      x, gamma2, beta2, gsum, gsumsq, out);
}
